// RegionEmbedder_39247411151462
// MI455X (gfx1250) — matrix operand order repaired and checked by reading, not compiled, not run on hardware
//
#include <hip/hip_runtime.h>
#include <hip/hip_bf16.h>

typedef __attribute__((ext_vector_type(16))) __bf16 v16bf;
typedef __attribute__((ext_vector_type(8)))  float  v8f;

#define N_NODES 50000
#define N_EDGES 800000
#define IN_DIM  512
#define HID_DIM 256
#define OUT_DIM 128
#define BN_EPS  1e-5f

struct U8 { unsigned int u[8]; };

__device__ __forceinline__ unsigned int f2bf(float f) {
  unsigned int u = __builtin_bit_cast(unsigned int, f);
  return (u + 0x7FFFu + ((u >> 16) & 1u)) >> 16;   // RNE truncate to bf16
}
__device__ __forceinline__ unsigned int pack2(float lo, float hi) {
  return f2bf(lo) | (f2bf(hi) << 16);
}

// ---------------------------------------------------------------------------
// Pack W [K,N] fp32 (row-major, x@W convention) into WMMA bf16 B-fragment
// order: P[((ntile*KB + kb)*32 + lane)*8 + j] = {W[k,col], W[k+1,col]} where
// col = ntile*16 + lane%16, k = kb*32 + (lane>=16 ? 16 : 0) + 2*j.
// ---------------------------------------------------------------------------
__global__ void gcn_pack_w(const float* __restrict__ W, unsigned int* __restrict__ P,
                           int K, int N) {
  int idx = blockIdx.x * blockDim.x + threadIdx.x;
  int KB = K / 32, NT = N / 16;
  if (idx >= NT * KB * 32 * 8) return;
  int j    = idx & 7;
  int rest = idx >> 3;
  int lane = rest & 31;  rest >>= 5;
  int kb   = rest % KB;
  int t    = rest / KB;
  int col  = t * 16 + (lane & 15);
  int k    = kb * 32 + ((lane & 16) ? 16 : 0) + 2 * j;
  P[idx] = pack2(W[k * N + col], W[(k + 1) * N + col]);
}

// ---------------------------------------------------------------------------
// C[M,N] = A[M,K](fp32) x Bpacked(bf16 fragments), fp32 accumulate via
// v_wmma_f32_16x16x32_bf16. One block = 16 rows; wave w owns N-tile w.
// A strip staged in LDS as bf16 (padded stride -> conflict-free ds_load_b128).
// M must be a multiple of 16 (50000 = 16*3125).
// ---------------------------------------------------------------------------
template <int K, int N>
__global__ void gcn_gemm_wmma(const float* __restrict__ A,
                              const unsigned int* __restrict__ Bp,
                              float* __restrict__ C, int M) {
  constexpr int KB   = K / 32;       // k-blocks of 32
  constexpr int LDSW = K / 2 + 4;    // padded dword stride per row
  __shared__ unsigned int lds[16 * LDSW];

  const int row0 = blockIdx.x * 16;
  const int tid  = threadIdx.x;

  // Stage 16 x K fp32 -> bf16x2 dwords in LDS (coalesced float2 reads).
  for (int i = tid; i < 16 * (K / 2); i += blockDim.x) {
    int r  = i / (K / 2);
    int c2 = i % (K / 2);
    const float2 f = reinterpret_cast<const float2*>(A + (size_t)(row0 + r) * K)[c2];
    lds[r * LDSW + c2] = pack2(f.x, f.y);
  }
  __syncthreads();

  const int wave = tid >> 5;
  const int lane = tid & 31;
  const int arow = lane & 15;
  const int ah   = (lane & 16) ? 8 : 0;   // dword offset for K+16 half

  v8f acc = {};
  for (int kb = 0; kb < KB; ++kb) {
    // A fragment: VGPR0-7 = K[kb*32 + (lane<16?0:16) .. +15].
    U8 a, b;
    const unsigned int* ab = &lds[arow * LDSW + kb * 16];
    *reinterpret_cast<uint4*>(&a.u[0]) = *reinterpret_cast<const uint4*>(ab + ah);
    *reinterpret_cast<uint4*>(&a.u[4]) = *reinterpret_cast<const uint4*>(ab + 4 + ah);
    // B fragment: fully coalesced 32B per lane.
    const unsigned int* bb = Bp + (size_t)(((wave * KB + kb) * 32 + lane)) * 8;
    *reinterpret_cast<uint4*>(&b.u[0]) = *reinterpret_cast<const uint4*>(bb);
    *reinterpret_cast<uint4*>(&b.u[4]) = *reinterpret_cast<const uint4*>(bb + 4);

    acc = __builtin_amdgcn_wmma_f32_16x16x32_bf16(
        false, __builtin_bit_cast(v16bf, a),
        false, __builtin_bit_cast(v16bf, b),
        (short)0, acc, false, false);
  }

  // D layout: VGPR v, lanes 0-15 -> M=v, lanes 16-31 -> M=8+v; col = lane%16.
  const int col   = wave * 16 + (lane & 15);
  const int rbase = row0 + ((lane & 16) ? 8 : 0);
#pragma unroll
  for (int v = 0; v < 8; ++v)
    C[(size_t)(rbase + v) * N + col] = acc[v];
}

// ---------------------------------------------------------------------------
// Edge scatter-add: one wave per edge, relaxed agent-scope f32 atomics (L2).
// ---------------------------------------------------------------------------
template <int CDIM>
__global__ void gcn_scatter(const float* __restrict__ H,
                            const long long* __restrict__ ei,
                            float* __restrict__ agg, int nEdges) {
  int e = blockIdx.x * (blockDim.x >> 5) + (threadIdx.x >> 5);
  if (e >= nEdges) return;
  int lane = threadIdx.x & 31;
  int s = (int)ei[e];
  int d = (int)ei[(size_t)nEdges + e];
  const float* hs = H + (size_t)s * CDIM;
  float* ad = agg + (size_t)d * CDIM;
#pragma unroll
  for (int c = lane; c < CDIM; c += 32)
    __hip_atomic_fetch_add(&ad[c], hs[c], __ATOMIC_RELAXED, __HIP_MEMORY_SCOPE_AGENT);
}

// ---------------------------------------------------------------------------
// BatchNorm column sums (thread c owns column c -> fully coalesced rows).
// ---------------------------------------------------------------------------
__global__ void gcn_bn_partial(const float* __restrict__ agg, float* __restrict__ sums,
                               int M) {
  int c = threadIdx.x;  // 0..HID_DIM-1
  float s = 0.f, ss = 0.f;
  for (int r = blockIdx.x; r < M; r += gridDim.x) {
    float v = agg[(size_t)r * HID_DIM + c];
    s += v; ss += v * v;
  }
  __hip_atomic_fetch_add(&sums[c],            s,  __ATOMIC_RELAXED, __HIP_MEMORY_SCOPE_AGENT);
  __hip_atomic_fetch_add(&sums[HID_DIM + c],  ss, __ATOMIC_RELAXED, __HIP_MEMORY_SCOPE_AGENT);
}

// Note: bias b1 shifts the mean and cancels inside BN, so stats use raw agg.
__global__ void gcn_bn_finalize(const float* __restrict__ sums,
                                const float* __restrict__ gamma,
                                const float* __restrict__ beta,
                                float* __restrict__ mst) {
  int c = threadIdx.x;
  float mean = sums[c] / (float)N_NODES;
  float var  = sums[HID_DIM + c] / (float)N_NODES - mean * mean;
  mst[c]               = mean;
  mst[HID_DIM + c]     = __frsqrt_rn(var + BN_EPS) * gamma[c];
  mst[2 * HID_DIM + c] = beta[c];
}

__global__ void gcn_bn_apply(const float* __restrict__ agg,
                             const float* __restrict__ mst,
                             float* __restrict__ Hb, int total) {
  for (int i = blockIdx.x * blockDim.x + threadIdx.x; i < total;
       i += gridDim.x * blockDim.x) {
    int c = i & (HID_DIM - 1);
    float v = (agg[i] - mst[c]) * mst[HID_DIM + c] + mst[2 * HID_DIM + c];
    Hb[i] = fmaxf(v, 0.f);
  }
}

// ---------------------------------------------------------------------------
// out = (agg2 + b2) / max(||row||, 1e-12).  One wave per row, shfl reduction.
// ---------------------------------------------------------------------------
__global__ void gcn_l2norm(const float* __restrict__ agg2,
                           const float* __restrict__ b2,
                           float* __restrict__ out, int M) {
  int r = blockIdx.x * (blockDim.x >> 5) + (threadIdx.x >> 5);
  if (r >= M) return;
  int lane = threadIdx.x & 31;
  float v[OUT_DIM / 32];
  float ss = 0.f;
#pragma unroll
  for (int j = 0; j < OUT_DIM / 32; ++j) {
    int c = lane + j * 32;
    v[j] = agg2[(size_t)r * OUT_DIM + c] + b2[c];
    ss += v[j] * v[j];
  }
#pragma unroll
  for (int off = 16; off > 0; off >>= 1) ss += __shfl_xor(ss, off, 32);
  float inv = 1.f / fmaxf(sqrtf(ss), 1e-12f);
#pragma unroll
  for (int j = 0; j < OUT_DIM / 32; ++j)
    out[(size_t)r * OUT_DIM + lane + j * 32] = v[j] * inv;
}

// ---------------------------------------------------------------------------
extern "C" void kernel_launch(void* const* d_in, const int* in_sizes, int n_in,
                              void* d_out, int out_size, void* d_ws, size_t ws_size,
                              hipStream_t stream) {
  const float*     x   = (const float*)d_in[0];
  const long long* ei  = (const long long*)d_in[1];   // int64 [2, N_EDGES]
  const float*     W1  = (const float*)d_in[2];
  // d_in[3] = b1 (cancels in BN, unused)
  const float*     gam = (const float*)d_in[4];
  const float*     bet = (const float*)d_in[5];
  const float*     W2  = (const float*)d_in[6];
  const float*     b2  = (const float*)d_in[7];
  float*           out = (float*)d_out;

  // Workspace layout (bytes). buf1: H1 -> Hb -> agg2. buf2: agg1 -> H2.
  const size_t SZ1 = (size_t)N_NODES * HID_DIM * sizeof(float);   // 51.2 MB
  char* ws   = (char*)d_ws;
  float* buf1 = (float*)ws;                       // [N_NODES, HID_DIM]
  float* buf2 = (float*)(ws + SZ1);               // [N_NODES, HID_DIM]
  unsigned int* P1 = (unsigned int*)(ws + 2 * SZ1);               // 256 KB
  unsigned int* P2 = (unsigned int*)(ws + 2 * SZ1 + 262144);      //  64 KB
  float* sums = (float*)(ws + 2 * SZ1 + 262144 + 65536);          // [2*HID]
  float* mst  = sums + 2 * HID_DIM;                               // [3*HID]

  // 1) Pack weights into WMMA B-fragment order (bf16).
  {
    int t1 = (IN_DIM / 32) * (HID_DIM / 16) * 256;   // 65536
    int t2 = (HID_DIM / 32) * (OUT_DIM / 16) * 256;  // 16384
    gcn_pack_w<<<(t1 + 255) / 256, 256, 0, stream>>>(W1, P1, IN_DIM, HID_DIM);
    gcn_pack_w<<<(t2 + 255) / 256, 256, 0, stream>>>(W2, P2, HID_DIM, OUT_DIM);
  }

  // 2) H1 = x @ W1  (bf16 WMMA, fp32 accumulate) -> buf1
  gcn_gemm_wmma<IN_DIM, HID_DIM><<<N_NODES / 16, (HID_DIM / 16) * 32, 0, stream>>>(
      x, P1, buf1, N_NODES);

  // 3) agg1 = segment_sum(H1[src], dst) -> buf2
  hipMemsetAsync(buf2, 0, SZ1, stream);
  gcn_scatter<HID_DIM><<<(N_EDGES + 7) / 8, 256, 0, stream>>>(buf1, ei, buf2, N_EDGES);

  // 4) BatchNorm stats + apply (+ReLU): Hb -> buf1 (overwrites H1)
  hipMemsetAsync(sums, 0, 2 * HID_DIM * sizeof(float), stream);
  gcn_bn_partial<<<256, HID_DIM, 0, stream>>>(buf2, sums, N_NODES);
  gcn_bn_finalize<<<1, HID_DIM, 0, stream>>>(sums, gam, bet, mst);
  gcn_bn_apply<<<2048, 256, 0, stream>>>(buf2, mst, buf1, N_NODES * HID_DIM);

  // 5) H2 = Hb @ W2 -> buf2 (overwrites agg1)
  gcn_gemm_wmma<HID_DIM, OUT_DIM><<<N_NODES / 16, (OUT_DIM / 16) * 32, 0, stream>>>(
      buf1, P2, buf2, N_NODES);

  // 6) agg2 = segment_sum(H2[src], dst) -> buf1 (overwrites Hb)
  hipMemsetAsync(buf1, 0, (size_t)N_NODES * OUT_DIM * sizeof(float), stream);
  gcn_scatter<OUT_DIM><<<(N_EDGES + 7) / 8, 256, 0, stream>>>(buf2, ei, buf1, N_EDGES);

  // 7) out = normalize(agg2 + b2, p=2, dim=1)
  gcn_l2norm<<<(N_NODES + 7) / 8, 256, 0, stream>>>(buf1, b2, out, N_NODES);
}
